// LAS_51247549776172
// MI455X (gfx1250) — compile-verified, matrix operand order checked by reading, not run on hardware
//
#include <hip/hip_runtime.h>

// ---------------------------------------------------------------- types
typedef __attribute__((ext_vector_type(16))) __bf16   v16bf;
typedef __attribute__((ext_vector_type(8)))  float    v8f;
typedef __attribute__((ext_vector_type(4)))  unsigned v4u;

#define DEV static __device__ __forceinline__

union FragU { v16bf v; v4u q[2]; };

DEV unsigned short f2bf(float x) {
  unsigned u = __float_as_uint(x);
  unsigned r = (u + 0x7FFFu + ((u >> 16) & 1u)) >> 16;
  return (unsigned short)r;
}
DEV float sigf(float x) { return 1.f / (1.f + __expf(-x)); }

// 16-bit A-fragment (16x32, M x K) per CDNA5 layout:
// lanes 0-15: row M=lane,    K = k0+{0..7, 16..23}
// lanes 16-31: row M=lane-16, K = k0+{8..15, 24..31}
// (B fragments use the same gather on row-major W = B^T.)
DEV v16bf frag_load(const unsigned short* base, size_t rstride, int r0, int k0, int lane) {
  int l = lane & 15, hi = lane >> 4;
  const unsigned short* p = base + (size_t)(r0 + l) * rstride + (size_t)k0 + (size_t)(hi * 8);
  FragU f;
  f.q[0] = *(const v4u*)(p);
  f.q[1] = *(const v4u*)(p + 16);
  return f.v;
}
DEV v16bf frag_load_clamp(const unsigned short* base, size_t rstride, int r0, int k0,
                          int lane, int rmax) {
  int l = lane & 15, hi = lane >> 4;
  int r = r0 + l; if (r >= rmax) r = rmax - 1;
  const unsigned short* p = base + (size_t)r * rstride + (size_t)k0 + (size_t)(hi * 8);
  FragU f;
  f.q[0] = *(const v4u*)(p);
  f.q[1] = *(const v4u*)(p + 16);
  return f.v;
}

DEV v8f wmma_bf16(v16bf a, v16bf b, v8f c) {
  return __builtin_amdgcn_wmma_f32_16x16x32_bf16(false, a, false, b, (short)0, c, false, false);
}

// device-wide software barrier for persistent recurrent kernels
DEV void grid_barrier(unsigned* cnt, unsigned* gen, int nwg) {
  __syncthreads();
  if (threadIdx.x == 0) {
    __threadfence();
    unsigned g   = __hip_atomic_load(gen, __ATOMIC_RELAXED, __HIP_MEMORY_SCOPE_AGENT);
    unsigned old = __hip_atomic_fetch_add(cnt, 1u, __ATOMIC_ACQ_REL, __HIP_MEMORY_SCOPE_AGENT);
    if (old == (unsigned)(nwg - 1)) {
      __hip_atomic_store(cnt, 0u, __ATOMIC_RELAXED, __HIP_MEMORY_SCOPE_AGENT);
      __hip_atomic_fetch_add(gen, 1u, __ATOMIC_ACQ_REL, __HIP_MEMORY_SCOPE_AGENT);
    } else {
      while (__hip_atomic_load(gen, __ATOMIC_ACQUIRE, __HIP_MEMORY_SCOPE_AGENT) == g) {
        __builtin_amdgcn_s_sleep(2);
      }
    }
    __threadfence();
  }
  __syncthreads();
}

// ---------------------------------------------------------------- small kernels
__global__ void cvt_bf16_k(const float* __restrict__ s, unsigned short* __restrict__ d, int n) {
  int i = blockIdx.x * blockDim.x + threadIdx.x;
  if (i < n) d[i] = f2bf(s[i]);
}

__global__ void combine_w_k(const float* __restrict__ Wih, const float* __restrict__ Whh,
                            unsigned short* __restrict__ dst, int N4, int Kin, int Hd) {
  int Kt = Kin + Hd;
  int i = blockIdx.x * blockDim.x + threadIdx.x;
  if (i >= N4 * Kt) return;
  int n = i / Kt, k = i % Kt;
  float v = (k < Kin) ? Wih[(size_t)n * Kin + k] : Whh[(size_t)n * Hd + (k - Kin)];
  dst[i] = f2bf(v);
}

// xs[b][tq][j*80+f] = x[b][tq*4+j][f] (bf16);  m[b][tq] = x_mask[b][tq*4][0]
__global__ void pack_x_k(const float* __restrict__ x, const float* __restrict__ xm,
                         unsigned short* __restrict__ xs, float* __restrict__ m) {
  int i = blockIdx.x * blockDim.x + threadIdx.x;
  if (i < 32 * 1024 * 320) {
    int b = i / (1024 * 320), r = i % (1024 * 320);
    int tq = r / 320, k = r % 320;
    int j = k / 80, f = k % 80;
    xs[i] = f2bf(x[((size_t)b * 4096 + tq * 4 + j) * 80 + f]);
  }
  if (i < 32 * 1024) m[i] = xm[(size_t)(i / 1024) * 4096 + (i % 1024) * 4];
}

// bl_bf[row][k] = (k<512 ? bf[row][k] : bb[row][k-512]) * m[row]
__global__ void concat_bl_k(const float* __restrict__ bfq, const float* __restrict__ bbq,
                            const float* __restrict__ m, unsigned short* __restrict__ d) {
  int i = blockIdx.x * blockDim.x + threadIdx.x;
  if (i >= 32 * 1024 * 1024) return;
  int row = i / 1024, k = i % 1024;
  float v = (k < 512) ? bfq[(size_t)row * 512 + k] : bbq[(size_t)row * 512 + k - 512];
  d[i] = f2bf(v * m[row]);
}

__global__ void embed_gather_k(const float* __restrict__ emb, const int* __restrict__ y,
                               unsigned short* __restrict__ d) {
  int i = blockIdx.x * blockDim.x + threadIdx.x;
  if (i >= 32 * 64 * 256) return;
  int b = i / (64 * 256), r = i % (64 * 256);
  int l = r / 256, p = r % 256;
  int tok = y[b * 65 + l];                                // y[:, :-1]
  d[i] = f2bf(emb[(size_t)tok * 256 + p]);
}

__global__ void attfea_k(const float* __restrict__ ls, const float* __restrict__ cs,
                         const float* __restrict__ ym, float* __restrict__ df,
                         unsigned short* __restrict__ db) {
  int i = blockIdx.x * blockDim.x + threadIdx.x;          // 32*64*512
  if (i >= 32 * 64 * 512) return;
  int row = i / 512, k = i % 512;
  float w = ym[(row / 64) * 65 + (row % 64) + 1];
  float v = ((k < 256) ? ls[(size_t)row * 256 + k] : cs[(size_t)row * 256 + k - 256]) * w;
  df[i] = v; db[i] = f2bf(v);
}

__global__ void dcat_k(const float* __restrict__ af, const float* __restrict__ dq,
                       const float* __restrict__ ym, unsigned short* __restrict__ d) {
  int i = blockIdx.x * blockDim.x + threadIdx.x;          // 32*64*768
  if (i >= 32 * 64 * 768) return;
  int row = i / 768, k = i % 768;
  float w = ym[(row / 64) * 65 + (row % 64) + 1];
  float v = (k < 512) ? af[(size_t)row * 512 + k] : dq[(size_t)row * 256 + k - 512];
  d[i] = f2bf(v * w);
}

// ---------------------------------------------------------------- WMMA GEMM (2x2 blocked)
// C[M,N] = act(A_bf16[M,K] @ W_bf16[N,K]^T + bias) * rowmask
// one 32x32 block per wave (4 accumulators): 1 load per WMMA.
__global__ void gemm_wmma_k(const unsigned short* __restrict__ A,
                            const unsigned short* __restrict__ W,
                            const float* __restrict__ bias,
                            float* Cf, unsigned short* Cbf,
                            int M, int N, int K, int act,
                            const float* __restrict__ rowmask) {
  int wave = threadIdx.x >> 5, lane = threadIdx.x & 31;
  int nbn = (N + 31) >> 5, nbm = M >> 5;
  int blk = blockIdx.x * 8 + wave;
  if (blk >= nbm * nbn) return;
  int bm = blk / nbn, bn = blk % nbn;
  int r0 = bm << 5, c0 = bn << 5;
  v8f acc[2][2] = {};
  for (int k0 = 0; k0 < K; k0 += 32) {
    v16bf a0 = frag_load(A, (size_t)K, r0, k0, lane);
    v16bf a1 = frag_load(A, (size_t)K, r0 + 16, k0, lane);
    v16bf b0 = frag_load_clamp(W, (size_t)K, c0, k0, lane, N);
    v16bf b1 = frag_load_clamp(W, (size_t)K, c0 + 16, k0, lane, N);
    acc[0][0] = wmma_bf16(a0, b0, acc[0][0]);
    acc[0][1] = wmma_bf16(a0, b1, acc[0][1]);
    acc[1][0] = wmma_bf16(a1, b0, acc[1][0]);
    acc[1][1] = wmma_bf16(a1, b1, acc[1][1]);
  }
  for (int i = 0; i < 2; ++i) {
    for (int j = 0; j < 2; ++j) {
      int col = c0 + j * 16 + (lane & 15);
      if (col >= N) continue;
      float bv = bias ? bias[col] : 0.f;
      for (int r = 0; r < 8; ++r) {
        int row = r0 + i * 16 + r + ((lane >> 4) << 3);
        float v = acc[i][j][r] + bv;
        if (act == 1) v = (v > 0.f) ? v : 0.1f * v;       // leaky_relu(0.1)
        if (rowmask) v *= rowmask[row];
        size_t o = (size_t)row * N + col;
        if (Cf)  Cf[o]  = v;
        if (Cbf) Cbf[o] = f2bf(v);
      }
    }
  }
}

// ---------------------------------------------------------------- pipelined encoder LSTM
// Role r works on step (round - delay): l1 -> l2 -> l3 -> blstm_f form a software
// pipeline over one device-wide barrier per round (1027 rounds instead of 4096).
// Each role: 16 WGs x 256 thr; WG owns 32 hidden cols; 8 waves = 4 gates x 2 col-tiles,
// each wave computes both batch m-tiles sharing one B fragment (weights read once,
// L2-resident). The 32x32 activation chunk is staged in LDS per k-block (ds_load_b128).
struct LRole {
  const unsigned short* in1;      // [32][1024][K1]
  const unsigned short* Wc;       // [2048][K1+512]
  const float* bias;              // [2048]
  float* out_f;                   // [32][1024][512] or null
  unsigned short* out_bf;         // [32][1024][512] or null
  const float* outmask;           // per (b,t) scale on outputs, or null
  unsigned short* h0; unsigned short* h1;   // bf16 [32][512] ping-pong
  float* c;                       // f32 [32][512]
  int K1; int delay; int reverse;
};

__global__ void enc_pipe_k(LRole R0, LRole R1, LRole R2, LRole R3,
                           int nroles, int rounds, unsigned* cnt, unsigned* gen) {
  LRole roles[4] = { R0, R1, R2, R3 };
  int role = blockIdx.x >> 4;
  LRole R = roles[role];
  int nwg = nroles << 4;
  int wg = blockIdx.x & 15;
  int j0 = wg << 5;
  int tid = threadIdx.x;
  int wave = tid >> 5, lane = tid & 31;
  int g = wave >> 1, ct = wave & 1;
  int K1 = R.K1, Kt = K1 + 512;
  int n0 = (g << 9) + j0 + (ct << 4);
  __shared__ unsigned short chunk[32][32];
  __shared__ float sg4[4][32][32];
  for (int rr = 0; rr < rounds; ++rr) {
    int s = rr - R.delay;
    if ((unsigned)s < 1024u) {
      int t = R.reverse ? (1023 - s) : s;
      const unsigned short* hr = ((s & 1) == 0) ? R.h0 : R.h1;
      unsigned short*       hw = ((s & 1) == 0) ? R.h1 : R.h0;
      v8f acc0 = {}, acc1 = {};
      for (int k0 = 0; k0 < Kt; k0 += 32) {
        {   // cooperative stage of A chunk (32 rows x 32 k, bf16)
          int row = tid >> 3, grp = tid & 7;
          int kk = k0 + (grp << 2);
          const unsigned short* src = (kk < K1)
              ? (R.in1 + ((size_t)row * 1024 + t) * K1 + kk)
              : (hr + (size_t)row * 512 + (kk - K1));
          *(unsigned long long*)&chunk[row][grp << 2] = *(const unsigned long long*)src;
        }
        __syncthreads();
        v16bf a0 = frag_load(&chunk[0][0], 32, 0, 0, lane);
        v16bf a1 = frag_load(&chunk[0][0], 32, 16, 0, lane);
        v16bf b  = frag_load(R.Wc, (size_t)Kt, n0, k0, lane);
        acc0 = wmma_bf16(a0, b, acc0);
        acc1 = wmma_bf16(a1, b, acc1);
        __syncthreads();
      }
      {   // gate tiles -> LDS (+bias)
        int col = lane & 15;
        float bv = R.bias[n0 + col];
        for (int r = 0; r < 8; ++r) {
          int row0 = r + ((lane >> 4) << 3);
          sg4[g][row0][(ct << 4) + col]      = acc0[r] + bv;
          sg4[g][row0 + 16][(ct << 4) + col] = acc1[r] + bv;
        }
      }
      __syncthreads();
      for (int e = tid; e < 1024; e += 256) {             // c/h update
        int b = e >> 5, j = e & 31;
        float iv = sg4[0][b][j], fv = sg4[1][b][j], gv = sg4[2][b][j], ov = sg4[3][b][j];
        int ci = b * 512 + j0 + j;
        float cn = sigf(fv) * R.c[ci] + sigf(iv) * tanhf(gv);
        float hn = sigf(ov) * tanhf(cn);
        R.c[ci] = cn;
        hw[ci]  = f2bf(hn);
        size_t so = ((size_t)b * 1024 + t) * 512 + j0 + j;
        float mk = R.outmask ? R.outmask[b * 1024 + t] : 1.f;
        if (R.out_f)  R.out_f[so]  = hn;
        if (R.out_bf) R.out_bf[so] = f2bf(hn * mk);
      }
      __syncthreads();
    }
    grid_barrier(cnt, gen, nwg);
  }
}

// ---------------------------------------------------------------- generic small LSTM (decoder)
// grid = Hd/16 WGs x 256 thr; wave (g,mt) computes one gate tile.
__global__ void lstm_wmma_k(const unsigned short* __restrict__ in1, int K1,
                            const unsigned short* __restrict__ in2, int K2,
                            int Hd, int Tq, int t_begin, int nsteps, int reverse,
                            const unsigned short* __restrict__ Wc,
                            const float* __restrict__ bias,
                            float* out_f, unsigned short* out_bf,
                            unsigned short* hA, unsigned short* hB, float* c_f,
                            unsigned* bar_cnt, unsigned* bar_gen, int nwg) {
  int wg = blockIdx.x;
  int wave = threadIdx.x >> 5, lane = threadIdx.x & 31;
  int g = wave >> 1, mt = wave & 1;
  int j0 = wg << 4;
  int n0 = g * Hd + j0;
  int Kt = K1 + K2 + Hd;
  size_t s1 = (size_t)Tq * K1;
  __shared__ float sg[4][32][16];
  for (int s = 0; s < nsteps; ++s) {
    int st = t_begin + s;
    int t  = reverse ? (Tq - 1 - st) : st;
    const unsigned short* hr = ((s & 1) == 0) ? hA : hB;
    unsigned short*       hw = ((s & 1) == 0) ? hB : hA;
    int r0 = mt << 4;
    v8f acc = {};
    for (int k0 = 0; k0 < Kt; k0 += 32) {
      v16bf a;
      if (k0 < K1)            a = frag_load(in1 + (size_t)t * K1, s1, r0, k0, lane);
      else if (k0 < K1 + K2)  a = frag_load(in2, (size_t)K2, r0, k0 - K1, lane);
      else                    a = frag_load(hr, (size_t)Hd, r0, k0 - K1 - K2, lane);
      v16bf b = frag_load(Wc, (size_t)Kt, n0, k0, lane);
      acc = wmma_bf16(a, b, acc);
    }
    int col = lane & 15;
    float bv = bias[n0 + col];
    for (int r = 0; r < 8; ++r) {
      int row = r0 + r + ((lane >> 4) << 3);
      sg[g][row][col] = acc[r] + bv;
    }
    __syncthreads();
    for (int e = threadIdx.x; e < 32 * 16; e += blockDim.x) {
      int b = e >> 4, j = e & 15;
      float iv = sg[0][b][j], fv = sg[1][b][j], gv = sg[2][b][j], ov = sg[3][b][j];
      int ci = b * Hd + j0 + j;
      float cn = sigf(fv) * c_f[ci] + sigf(iv) * tanhf(gv);
      float hn = sigf(ov) * tanhf(cn);
      c_f[ci] = cn;
      hw[ci]  = f2bf(hn);
      size_t so = ((size_t)b * Tq + t) * Hd + j0 + j;
      if (out_f)  out_f[so]  = hn;
      if (out_bf) out_bf[so] = f2bf(hn);
    }
    if (nsteps > 1) grid_barrier(bar_cnt, bar_gen, nwg);
    else __syncthreads();
  }
}

// ---------------------------------------------------------------- attention step
__global__ void attn_k(const float* __restrict__ state, const float* __restrict__ att_h,
                       const float* __restrict__ eout, const float* __restrict__ m,
                       const float* __restrict__ wv,
                       float* __restrict__ ctxs_seq, unsigned short* __restrict__ ctx_bf,
                       int step) {
  int b = blockIdx.x, tid = threadIdx.x;
  __shared__ float sc[1024];
  __shared__ float red[256];
  __shared__ float sst[256];
  __shared__ float swv[256];
  sst[tid] = state[b * 256 + tid];
  swv[tid] = wv[tid];
  __syncthreads();
  for (int t = tid; t < 1024; t += 256) {
    const float* ah = att_h + ((size_t)b * 1024 + t) * 256;
    float a = 0.f;
    for (int p = 0; p < 256; ++p) a += swv[p] * tanhf(sst[p] + ah[p]);
    sc[t] = a + (m[b * 1024 + t] - 1.f) * 1e30f;
  }
  __syncthreads();
  float mx = -3.4e38f;
  for (int t = tid; t < 1024; t += 256) mx = fmaxf(mx, sc[t]);
  red[tid] = mx; __syncthreads();
  for (int o = 128; o > 0; o >>= 1) { if (tid < o) red[tid] = fmaxf(red[tid], red[tid + o]); __syncthreads(); }
  mx = red[0]; __syncthreads();
  float ss = 0.f;
  for (int t = tid; t < 1024; t += 256) { float e = __expf(sc[t] - mx); sc[t] = e; ss += e; }
  red[tid] = ss; __syncthreads();
  for (int o = 128; o > 0; o >>= 1) { if (tid < o) red[tid] += red[tid + o]; __syncthreads(); }
  float inv = 1.f / red[0];
  __syncthreads();
  float a = 0.f;
  for (int t = 0; t < 1024; ++t) a += eout[((size_t)b * 1024 + t) * 256 + tid] * sc[t];
  a *= inv;
  ctxs_seq[((size_t)b * 64 + step) * 256 + tid] = a;
  ctx_bf[b * 256 + tid] = f2bf(a);
}

// ---------------------------------------------------------------- loss
__global__ void loss_k(const float* __restrict__ logits, const int* __restrict__ y,
                       const float* __restrict__ ym, float* acc) {
  int row = blockIdx.x, tid = threadIdx.x;
  int b = row >> 6, l = row & 63;
  const float* lr = logits + (size_t)row * 4235;
  __shared__ float red[256];
  float mx = -3.4e38f;
  for (int v = tid; v < 4235; v += 256) mx = fmaxf(mx, lr[v]);
  red[tid] = mx; __syncthreads();
  for (int o = 128; o > 0; o >>= 1) { if (tid < o) red[tid] = fmaxf(red[tid], red[tid + o]); __syncthreads(); }
  mx = red[0]; __syncthreads();
  float se = 0.f, sl = 0.f;
  for (int v = tid; v < 4235; v += 256) { se += __expf(lr[v] - mx); sl += lr[v]; }
  red[tid] = se; __syncthreads();
  for (int o = 128; o > 0; o >>= 1) { if (tid < o) red[tid] += red[tid + o]; __syncthreads(); }
  float set = red[0]; __syncthreads();
  red[tid] = sl; __syncthreads();
  for (int o = 128; o > 0; o >>= 1) { if (tid < o) red[tid] += red[tid + o]; __syncthreads(); }
  float slt = red[0];
  if (tid == 0) {
    float lse = mx + logf(set);
    float mk  = (ym[b * 65 + l + 1] > 0.f) ? 1.f : 0.f;
    int lab   = y[b * 65 + l + 1];
    atomicAdd(&acc[0], (4235.f * lse - slt) * mk);
    atomicAdd(&acc[1], (lse - lr[lab]) * mk);
    atomicAdd(&acc[2], mk);
  }
}

__global__ void final_k(const float* acc, float* out) {
  float n = acc[2];
  out[0] = (acc[0] / n) * (0.1f / 4235.f) + 0.9f * (acc[1] / n);
}

// ---------------------------------------------------------------- host
extern "C" void kernel_launch(void* const* d_in, const int* in_sizes, int n_in,
                              void* d_out, int out_size, void* d_ws, size_t ws_size,
                              hipStream_t stream) {
  const float* x      = (const float*)d_in[0];
  const float* x_mask = (const float*)d_in[1];
  const int*   y      = (const int*)d_in[2];
  const float* y_mask = (const float*)d_in[3];
  const float *l1Wih=(const float*)d_in[4],  *l1Whh=(const float*)d_in[5],  *l1b=(const float*)d_in[6];
  const float *l2Wih=(const float*)d_in[7],  *l2Whh=(const float*)d_in[8],  *l2b=(const float*)d_in[9];
  const float *l3Wih=(const float*)d_in[10], *l3Whh=(const float*)d_in[11], *l3b=(const float*)d_in[12];
  const float *bfWih=(const float*)d_in[13], *bfWhh=(const float*)d_in[14], *bfb=(const float*)d_in[15];
  const float *bbWih=(const float*)d_in[16], *bbWhh=(const float*)d_in[17], *bbb=(const float*)d_in[18];
  const float *c1W=(const float*)d_in[19], *c1b=(const float*)d_in[20];
  const float *c2W=(const float*)d_in[21], *c2b=(const float*)d_in[22];
  const float *emb=(const float*)d_in[23];
  const float *aWih=(const float*)d_in[24], *aWhh=(const float*)d_in[25], *ab=(const float*)d_in[26];
  const float *wW=(const float*)d_in[27], *wb=(const float*)d_in[28];
  const float *vW=(const float*)d_in[29], *vb=(const float*)d_in[30];
  const float *wav=(const float*)d_in[31];
  const float *dWih=(const float*)d_in[32], *dWhh=(const float*)d_in[33], *db=(const float*)d_in[34];
  const float *clsW=(const float*)d_in[35], *clsb=(const float*)d_in[36];

  size_t off = 0;
  auto alloc = [&](size_t bytes) -> void* {
    off = (off + 255) & ~(size_t)255;
    void* p = (char*)d_ws + off; off += bytes; return p;
  };
  typedef unsigned short us;

  // bf16 weights
  us* wc1   = (us*)alloc((size_t)2048 * 832 * 2);
  us* wc2   = (us*)alloc((size_t)2048 * 1024 * 2);
  us* wc3   = (us*)alloc((size_t)2048 * 1024 * 2);
  us* wcbf  = (us*)alloc((size_t)2048 * 1024 * 2);
  us* wcbb  = (us*)alloc((size_t)2048 * 1024 * 2);
  us* wcatt = (us*)alloc((size_t)1024 * 768 * 2);
  us* wcdec = (us*)alloc((size_t)1024 * 768 * 2);
  us* c1Wb  = (us*)alloc((size_t)512 * 1024 * 2);
  us* c2Wb  = (us*)alloc((size_t)256 * 512 * 2);
  us* wWb   = (us*)alloc((size_t)256 * 256 * 2);
  us* vWb   = (us*)alloc((size_t)256 * 256 * 2);
  us* clsWb = (us*)alloc((size_t)4235 * 768 * 2);
  // activations
  us*    xs   = (us*)alloc((size_t)32 * 1024 * 320 * 2);
  float* m    = (float*)alloc((size_t)32 * 1024 * 4);
  us*    S0b  = (us*)alloc((size_t)32 * 1024 * 512 * 2);   // l1 out
  us*    S1b  = (us*)alloc((size_t)32 * 1024 * 512 * 2);   // l2 out
  us*    S2b  = (us*)alloc((size_t)32 * 1024 * 512 * 2);   // l3 out * m
  float* S1f  = (float*)alloc((size_t)32 * 1024 * 512 * 4); // blstm_f out
  float* S2f  = (float*)alloc((size_t)32 * 1024 * 512 * 4); // blstm_b out
  us*    blb  = (us*)alloc((size_t)32 * 1024 * 1024 * 2);
  us*    c1o  = (us*)alloc((size_t)32 * 1024 * 512 * 2);
  float* eout = (float*)alloc((size_t)32 * 1024 * 256 * 4);
  us*    eoutb= (us*)alloc((size_t)32 * 1024 * 256 * 2);
  float* atth = (float*)alloc((size_t)32 * 1024 * 256 * 4);
  // encoder recurrent state: 5 roles x (2 h ping-pong bf16 + c f32)
  us*    hst  = (us*)alloc((size_t)10 * 32 * 512 * 2);
  float* cst  = (float*)alloc((size_t)5 * 32 * 512 * 4);
  // decoder
  us*    embb = (us*)alloc((size_t)32 * 64 * 256 * 2);
  float* lstms= (float*)alloc((size_t)32 * 64 * 256 * 4);
  float* ctxs = (float*)alloc((size_t)32 * 64 * 256 * 4);
  float* state= (float*)alloc((size_t)32 * 256 * 4);
  us*    ctxb = (us*)alloc((size_t)32 * 256 * 2);
  us*    hA   = (us*)alloc((size_t)32 * 512 * 2);
  us*    hB   = (us*)alloc((size_t)32 * 512 * 2);
  float* cbuf = (float*)alloc((size_t)32 * 512 * 4);
  float* aff  = (float*)alloc((size_t)32 * 64 * 512 * 4);
  us*    affb = (us*)alloc((size_t)32 * 64 * 512 * 2);
  float* dout = (float*)alloc((size_t)32 * 64 * 256 * 4);
  us*    dcat = (us*)alloc((size_t)32 * 64 * 768 * 2);
  float* logit= (float*)alloc((size_t)2048 * 4235 * 4);
  float* acc  = (float*)alloc(4 * 4);
  unsigned* bar = (unsigned*)alloc(2 * 4);

  auto cvt = [&](const float* s, us* d, int n) {
    cvt_bf16_k<<<dim3((n + 255) / 256), dim3(256), 0, stream>>>(s, d, n);
  };
  auto comb = [&](const float* Wih, const float* Whh, us* d, int N4, int Kin, int Hd) {
    int n = N4 * (Kin + Hd);
    combine_w_k<<<dim3((n + 255) / 256), dim3(256), 0, stream>>>(Wih, Whh, d, N4, Kin, Hd);
  };
  auto gemm = [&](const us* A, const us* W, const float* bias, float* Cf, us* Cbf,
                  int M, int N, int K, int act, const float* rmask) {
    int blocks = (M >> 5) * ((N + 31) >> 5);
    gemm_wmma_k<<<dim3((blocks + 7) / 8), dim3(256), 0, stream>>>(A, W, bias, Cf, Cbf,
                                                                  M, N, K, act, rmask);
  };
  auto role = [&](const us* in1, int K1, const us* Wc, const float* bias,
                  float* of, us* ob, const float* omask, int ri, int delay, int rev) {
    LRole r;
    r.in1 = in1; r.Wc = Wc; r.bias = bias;
    r.out_f = of; r.out_bf = ob; r.outmask = omask;
    r.h0 = hst + (size_t)(2 * ri) * 32 * 512;
    r.h1 = hst + (size_t)(2 * ri + 1) * 32 * 512;
    r.c  = cst + (size_t)ri * 32 * 512;
    r.K1 = K1; r.delay = delay; r.reverse = rev;
    return r;
  };
  auto lstm = [&](const us* in1, int K1, const us* in2, int K2, int Hd, int Tq,
                  int t0, int ns, int rev, const us* Wc, const float* bias,
                  float* of, us* ob, us* ha, us* hb) {
    lstm_wmma_k<<<dim3(Hd / 16), dim3(256), 0, stream>>>(in1, K1, in2, K2, Hd, Tq, t0, ns,
                                                         rev, Wc, bias, of, ob, ha, hb,
                                                         cbuf, bar, bar + 1, Hd / 16);
  };

  // weights -> bf16
  comb(l1Wih, l1Whh, wc1, 2048, 320, 512);
  comb(l2Wih, l2Whh, wc2, 2048, 512, 512);
  comb(l3Wih, l3Whh, wc3, 2048, 512, 512);
  comb(bfWih, bfWhh, wcbf, 2048, 512, 512);
  comb(bbWih, bbWhh, wcbb, 2048, 512, 512);
  comb(aWih, aWhh, wcatt, 1024, 512, 256);
  comb(dWih, dWhh, wcdec, 1024, 512, 256);
  cvt(c1W, c1Wb, 512 * 1024); cvt(c2W, c2Wb, 256 * 512);
  cvt(wW, wWb, 256 * 256);    cvt(vW, vWb, 256 * 256);
  cvt(clsW, clsWb, 4235 * 768);

  // pack input + mask
  pack_x_k<<<dim3((32 * 1024 * 320 + 255) / 256), dim3(256), 0, stream>>>(x, x_mask, xs, m);

  // encoder: pipelined l1 -> l2 -> l3(masked out) -> blstm_f   (1027 barrier rounds)
  hipMemsetAsync(hst, 0, (size_t)10 * 32 * 512 * 2, stream);
  hipMemsetAsync(cst, 0, (size_t)5 * 32 * 512 * 4, stream);
  hipMemsetAsync(bar, 0, 8, stream);
  {
    LRole r0 = role(xs,  320, wc1,  l1b, nullptr, S0b, nullptr, 0, 0, 0);
    LRole r1 = role(S0b, 512, wc2,  l2b, nullptr, S1b, nullptr, 1, 1, 0);
    LRole r2 = role(S1b, 512, wc3,  l3b, nullptr, S2b, m,       2, 2, 0);
    LRole r3 = role(S2b, 512, wcbf, bfb, S1f,  nullptr, nullptr, 3, 3, 0);
    enc_pipe_k<<<dim3(64), dim3(256), 0, stream>>>(r0, r1, r2, r3, 4, 1027, bar, bar + 1);
  }
  // blstm backward (reverse time; cannot join forward pipeline)
  hipMemsetAsync(bar, 0, 8, stream);
  {
    LRole r4 = role(S2b, 512, wcbb, bbb, S2f, nullptr, nullptr, 4, 0, 1);
    enc_pipe_k<<<dim3(16), dim3(256), 0, stream>>>(r4, r4, r4, r4, 1, 1024, bar, bar + 1);
  }
  concat_bl_k<<<dim3((32 * 1024 * 1024 + 255) / 256), dim3(256), 0, stream>>>(S1f, S2f, m, blb);

  // conv1 (leaky) / conv2 (masked) / att_h projection
  gemm(blb, c1Wb, c1b, nullptr, c1o, 32768, 512, 1024, 1, nullptr);
  gemm(c1o, c2Wb, c2b, eout, eoutb, 32768, 256, 512, 0, m);
  gemm(eoutb, wWb, wb, atth, nullptr, 32768, 256, 256, 0, nullptr);

  // decoder
  embed_gather_k<<<dim3((32 * 64 * 256 + 255) / 256), dim3(256), 0, stream>>>(emb, y, embb);
  hipMemsetAsync(hA, 0, (size_t)32 * 512 * 2, stream);
  hipMemsetAsync(hB, 0, (size_t)32 * 512 * 2, stream);
  hipMemsetAsync(cbuf, 0, (size_t)32 * 512 * 4, stream);
  hipMemsetAsync(ctxb, 0, (size_t)32 * 256 * 2, stream);
  hipMemsetAsync(bar, 0, 8, stream);
  for (int step = 0; step < 64; ++step) {
    us* hr = (step & 1) ? hB : hA;
    us* hw = (step & 1) ? hA : hB;
    lstm(embb, 256, ctxb, 256, 256, 64, step, 1, 0, wcatt, ab, lstms, nullptr, hr, hw);
    gemm(hw, vWb, vb, state, nullptr, 32, 256, 256, 0, nullptr);
    attn_k<<<dim3(32), dim3(256), 0, stream>>>(state, atth, eout, m, wav, ctxs, ctxb, step);
  }
  attfea_k<<<dim3((32 * 64 * 512 + 255) / 256), dim3(256), 0, stream>>>(lstms, ctxs, y_mask, aff, affb);
  hipMemsetAsync(hA, 0, (size_t)32 * 512 * 2, stream);
  hipMemsetAsync(hB, 0, (size_t)32 * 512 * 2, stream);
  hipMemsetAsync(cbuf, 0, (size_t)32 * 512 * 4, stream);
  hipMemsetAsync(bar, 0, 8, stream);
  lstm(affb, 512, nullptr, 0, 256, 64, 0, 64, 0, wcdec, db, dout, nullptr, hA, hB);
  dcat_k<<<dim3((32 * 64 * 768 + 255) / 256), dim3(256), 0, stream>>>(aff, dout, y_mask, dcat);
  gemm(dcat, clsWb, clsb, logit, nullptr, 2048, 4235, 768, 0, nullptr);

  // loss
  hipMemsetAsync(acc, 0, 16, stream);
  loss_k<<<dim3(2048), dim3(256), 0, stream>>>(logit, y, y_mask, acc);
  final_k<<<dim3(1), dim3(1), 0, stream>>>(acc, (float*)d_out);
}
